// EMAVectorQuantizer_8641474200353
// MI455X (gfx1250) — matrix operand order repaired and checked by reading, not compiled, not run on hardware
//
#include <hip/hip_runtime.h>
#include <hip/hip_bf16.h>

// ---------------------------------------------------------------------------
// EMA Vector Quantizer for MI455X (gfx1250, wave32, WMMA bf16 16x16x32)
//
// N = 65536 rows, D = 256, K = 1024 codes.
//   dist GEMM : flat[N,D] @ cb[K,D]^T  via v_wmma_f32_16x16x32_bf16
//               codebook tiles staged to LDS with async global->LDS copies
//               (ASYNCcnt) and double-buffered so the fetch of tile ct+1
//               overlaps the WMMA chain of tile ct.
//   argmin    : per-row running min in C-fragment layout + LDS reduction
//   outputs   : quantized, one-hot encodings, cs, new_ema_w, new_codebook
// ---------------------------------------------------------------------------

typedef __attribute__((ext_vector_type(16))) __bf16 v16bf;
typedef __attribute__((ext_vector_type(8)))  float  v8f;
typedef __attribute__((ext_vector_type(4)))  int    v4i;

#define NROWS   65536
#define DIM     256
#define NCODE   1024
#define DECAY_F 0.99f
#define OMD_F   0.01f
#define EPS_F   1e-5f

// d_out layout (floats)
#define Q_OFF   0u
#define E_OFF   16777216u
#define CS_OFF  83886080u
#define W_OFF   83887104u
#define CB_OFF  84149248u

// ---- CDNA5 async global->LDS copy (16B per lane), ASYNCcnt-tracked --------
typedef __attribute__((address_space(1))) v4i gv4i;   // global ptr elem type
typedef __attribute__((address_space(3))) v4i lv4i;   // LDS ptr elem type

__device__ __forceinline__ void async_copy16(const void* gsrc, void* ldst) {
#if __has_builtin(__builtin_amdgcn_global_load_async_to_lds_b128)
    __builtin_amdgcn_global_load_async_to_lds_b128(
        (gv4i*)(unsigned long long)gsrc,
        (lv4i*)(unsigned int)(unsigned long long)ldst, 0, 0);
#else
    // VDST = LDS byte offset (low 32 bits of the flat shared address)
    const unsigned int loff = (unsigned int)(unsigned long long)ldst;
    asm volatile("global_load_async_to_lds_b128 %0, %1, off"
                 :: "v"(loff), "v"(gsrc) : "memory");
#endif
}

__device__ __forceinline__ void wait_async0() {
#if __has_builtin(__builtin_amdgcn_s_wait_asynccnt)
    __builtin_amdgcn_s_wait_asynccnt(0);
#else
    asm volatile("s_wait_asynccnt 0x0" ::: "memory");
#endif
}

// ---------------------------------------------------------------------------
// Prep: codebook fp32 -> bf16, per-code squared norm, zero dw/counts.
// ---------------------------------------------------------------------------
__global__ __launch_bounds__(256) void vq_prep(
    const float* __restrict__ codebook,
    __bf16* __restrict__ cb16,
    float* __restrict__ cnorm,
    float* __restrict__ counts,
    float* __restrict__ dw)
{
    __shared__ float red[256];
    const int k = blockIdx.x;
    const int t = threadIdx.x;
    const int i = k * DIM + t;
    const float c = codebook[i];
    cb16[i] = (__bf16)c;
    dw[i]   = 0.0f;
    red[t]  = c * c;
    __syncthreads();
    #pragma unroll
    for (int s = 128; s > 0; s >>= 1) {
        if (t < s) red[t] += red[t + s];
        __syncthreads();
    }
    if (t == 0) { cnorm[k] = red[0]; counts[k] = 0.0f; }
}

// ---------------------------------------------------------------------------
// Main kernel: distances + argmin + all per-row outputs.
// grid = 512 blocks, block = 256 threads = 8 waves; each wave owns 16 rows.
// ---------------------------------------------------------------------------
__global__ __launch_bounds__(256) void vq_main(
    const float* __restrict__ flat,       // [N][D] fp32
    const float* __restrict__ codebook,   // [K][D] fp32 (for quantized gather)
    const __bf16* __restrict__ cb16,      // [K][D] bf16
    const float* __restrict__ cnorm,      // [K]
    float* __restrict__ counts,           // [K]
    float* __restrict__ dw,               // [K][D]
    float* __restrict__ quant,            // [N][D]
    float* __restrict__ enc)              // [N][K]
{
    __shared__ __align__(32) __bf16 sB[2][16 * DIM];  // 16 KB double-buffered tile
    __shared__ float sRedD[8 * 16 * 16];              // 8 KB: [wave][row][slot]
    __shared__ int   sRedI[8 * 16 * 16];              // 8 KB
    __shared__ int   sIdx[8 * 16];                    // winning code per (wave,row)

    const int tid  = threadIdx.x;
    const int wave = tid >> 5;
    const int lane = tid & 31;
    const int r    = lane & 15;       // A: row M; B: column N; C: column N
    const int h    = lane >> 4;       // lane half
    const int rowBase = blockIdx.x * 128 + wave * 16;

    // per-thread staging slot: code row (tid>>4), 16-dim segment (tid&15)
    const int stCode = tid >> 4;
    const int stDof  = (tid & 15) * 16;

    // ---- Preload A fragments for this wave's 16 rows (contiguous k order, same as B) --
    // lane (r,h): element e -> K = f*32 + 16h + e
    v16bf A[8];
    {
        const float* rp = flat + (size_t)(rowBase + r) * DIM;
        #pragma unroll
        for (int f = 0; f < 8; ++f) {
            const float4 x0 = *(const float4*)(rp + f * 32 + 16 * h);
            const float4 x1 = *(const float4*)(rp + f * 32 + 16 * h + 4);
            const float4 y0 = *(const float4*)(rp + f * 32 + 16 * h + 8);
            const float4 y1 = *(const float4*)(rp + f * 32 + 16 * h + 12);
            A[f][0]  = (__bf16)x0.x;  A[f][1]  = (__bf16)x0.y;
            A[f][2]  = (__bf16)x0.z;  A[f][3]  = (__bf16)x0.w;
            A[f][4]  = (__bf16)x1.x;  A[f][5]  = (__bf16)x1.y;
            A[f][6]  = (__bf16)x1.z;  A[f][7]  = (__bf16)x1.w;
            A[f][8]  = (__bf16)y0.x;  A[f][9]  = (__bf16)y0.y;
            A[f][10] = (__bf16)y0.z;  A[f][11] = (__bf16)y0.w;
            A[f][12] = (__bf16)y1.x;  A[f][13] = (__bf16)y1.y;
            A[f][14] = (__bf16)y1.z;  A[f][15] = (__bf16)y1.w;
        }
    }

    // Running (min,argmin); C element v -> row M = v + 8h, col = code-tile + r.
    float bestD[8];
    int   bestI[8];
    #pragma unroll
    for (int e = 0; e < 8; ++e) { bestD[e] = 3.4e38f; bestI[e] = 0; }

    // ---- Prologue: async-stage tile 0 into buffer 0 -------------------------
    {
        const uint4* src = (const uint4*)(cb16 + (size_t)stCode * DIM + stDof);
        uint4* dst = (uint4*)(sB[0] + stCode * DIM + stDof);
        async_copy16(src + 0, dst + 0);
        async_copy16(src + 1, dst + 1);
    }
    wait_async0();
    __syncthreads();

    // ---- Sweep the 1024 codes in 64 tiles of 16 -----------------------------
    for (int ct = 0; ct < 64; ++ct) {
        const int b = ct & 1;

        // async prefetch of tile ct+1 into the alternate buffer; overlaps WMMA
        if (ct + 1 < 64) {
            const uint4* src = (const uint4*)
                (cb16 + (size_t)((ct + 1) * 16 + stCode) * DIM + stDof);
            uint4* dst = (uint4*)(sB[b ^ 1] + stCode * DIM + stDof);
            async_copy16(src + 0, dst + 0);
            async_copy16(src + 1, dst + 1);
        }

        const float cn = cnorm[ct * 16 + r];   // L2-hot; latency hides under WMMA

        v8f acc = {};
        #pragma unroll
        for (int f = 0; f < 8; ++f) {
            // B layout: lane (r,h), element e -> K = f*32 + 16h + e, col N = r
            const v16bf B = *(const v16bf*)(sB[b] + r * DIM + f * 32 + 16 * h);
            acc = __builtin_amdgcn_wmma_f32_16x16x32_bf16(
                false, A[f], false, B, (short)0, acc, false, false);
        }

        const int code = ct * 16 + r;
        #pragma unroll
        for (int e = 0; e < 8; ++e) {
            const float d2 = cn - 2.0f * acc[e];   // ||c||^2 - 2 f.c (row norm dropped)
            if (d2 < bestD[e]) { bestD[e] = d2; bestI[e] = code; }  // codes ascend -> first-min
        }

        wait_async0();      // this wave's prefetch landed in LDS
        __syncthreads();    // everyone's prefetch landed; everyone's reads done
    }

    // ---- Cross-lane argmin via LDS transpose + ordered scan -----------------
    {
        float* myD = sRedD + wave * 256;
        int*   myI = sRedI + wave * 256;
        #pragma unroll
        for (int e = 0; e < 8; ++e) {
            myD[(e + 8 * h) * 16 + r] = bestD[e];
            myI[(e + 8 * h) * 16 + r] = bestI[e];
        }
    }
    __syncthreads();
    if (lane < 16) {
        const float* myD = sRedD + wave * 256 + lane * 16;
        const int*   myI = sRedI + wave * 256 + lane * 16;
        float bd = myD[0];
        int   bi = myI[0];
        #pragma unroll
        for (int s = 1; s < 16; ++s) {
            const float d = myD[s];
            const int   i = myI[s];
            if (d < bd || (d == bd && i < bi)) { bd = d; bi = i; }
        }
        sIdx[wave * 16 + lane] = bi;
        atomicAdd(&counts[bi], 1.0f);   // integer-valued f32: exact & order-free
    }
    __syncthreads();

    // ---- Per-row outputs: one-hot encodings, quantized gather, dw scatter ---
    for (int rr = 0; rr < 16; ++rr) {
        const int    idxv = sIdx[wave * 16 + rr];
        const size_t n    = (size_t)rowBase + rr;

        // encodings row: 1024 floats, coalesced float4 stores
        float* ep = enc + n * NCODE;
        #pragma unroll
        for (int j = 0; j < 8; ++j) {
            const int c0 = j * 128 + lane * 4;
            float4 v;
            v.x = (c0 + 0 == idxv) ? 1.0f : 0.0f;
            v.y = (c0 + 1 == idxv) ? 1.0f : 0.0f;
            v.z = (c0 + 2 == idxv) ? 1.0f : 0.0f;
            v.w = (c0 + 3 == idxv) ? 1.0f : 0.0f;
            *(float4*)(ep + c0) = v;
        }

        // quantized row: gather fp32 codebook[idxv]
        const float* cbp = codebook + (size_t)idxv * DIM;
        float* qp = quant + n * DIM;
        #pragma unroll
        for (int j = 0; j < 2; ++j) {
            const int c0 = j * 128 + lane * 4;
            *(float4*)(qp + c0) = *(const float4*)(cbp + c0);
        }

        // dw[idxv] += flat[n]  (scatter-accumulate, L2-resident target)
        const float* fp2 = flat + n * DIM;
        float* dwp = dw + (size_t)idxv * DIM;
        #pragma unroll
        for (int j = 0; j < 2; ++j) {
            const int c0 = j * 128 + lane * 4;
            const float4 a = *(const float4*)(fp2 + c0);
            atomicAdd(dwp + c0 + 0, a.x);
            atomicAdd(dwp + c0 + 1, a.y);
            atomicAdd(dwp + c0 + 2, a.z);
            atomicAdd(dwp + c0 + 3, a.w);
        }
    }
}

// ---------------------------------------------------------------------------
// cs kernel: single 1024-thread block, deterministic LDS tree reduction.
// ---------------------------------------------------------------------------
__global__ __launch_bounds__(1024) void vq_cs(
    const float* __restrict__ counts,
    const float* __restrict__ ema_cs,
    float* __restrict__ cs_out)
{
    __shared__ float red[1024];
    const int t = threadIdx.x;
    const float cs = ema_cs[t] * DECAY_F + OMD_F * counts[t];
    red[t] = cs;
    __syncthreads();
    #pragma unroll
    for (int s = 512; s > 0; s >>= 1) {
        if (t < s) red[t] += red[t + s];
        __syncthreads();
    }
    const float ntot = red[0];
    cs_out[t] = (cs + EPS_F) / (ntot + cs * EPS_F) * ntot;
}

// ---------------------------------------------------------------------------
// Finalize: new_ema_w = ema_w*DECAY + (1-DECAY)*dw ; new_codebook = w / cs[k]
// ---------------------------------------------------------------------------
__global__ __launch_bounds__(256) void vq_final(
    const float* __restrict__ ema_w,
    const float* __restrict__ dw,
    const float* __restrict__ cs,
    float* __restrict__ out_w,
    float* __restrict__ out_cb)
{
    const int i = blockIdx.x * 256 + threadIdx.x;   // 262144 elements
    const int k = i >> 8;
    const float w = ema_w[i] * DECAY_F + OMD_F * dw[i];
    out_w[i]  = w;
    out_cb[i] = w / cs[k];
}

// ---------------------------------------------------------------------------
extern "C" void kernel_launch(void* const* d_in, const int* in_sizes, int n_in,
                              void* d_out, int out_size, void* d_ws, size_t ws_size,
                              hipStream_t stream) {
    const float* flat     = (const float*)d_in[0];   // inputs [16,64,64,256]
    const float* codebook = (const float*)d_in[1];   // [1024,256]
    const float* ema_w    = (const float*)d_in[2];   // [1024,256]
    const float* ema_cs   = (const float*)d_in[3];   // [1024]
    float* out = (float*)d_out;

    // workspace layout
    char* ws = (char*)d_ws;
    __bf16* cb16  = (__bf16*)(ws);                   // 512 KB
    float*  cnorm = (float*)(ws + 524288);           // 4 KB
    float*  cnts  = (float*)(ws + 528384);           // 4 KB
    float*  dw    = (float*)(ws + 532480);           // 1 MB

    vq_prep <<<NCODE, 256, 0, stream>>>(codebook, cb16, cnorm, cnts, dw);
    vq_main <<<NROWS / 128, 256, 0, stream>>>(flat, codebook, cb16, cnorm, cnts, dw,
                                              out + Q_OFF, out + E_OFF);
    vq_cs   <<<1, 1024, 0, stream>>>(cnts, ema_cs, out + CS_OFF);
    vq_final<<<NCODE, 256, 0, stream>>>(ema_w, dw, out + CS_OFF, out + W_OFF, out + CB_OFF);
}
